// ClassificationNet_11269994184931
// MI455X (gfx1250) — compile-verified, matrix operand order checked by reading, not run on hardware
//
#include <hip/hip_runtime.h>
#include <hip/hip_bf16.h>
#include <math.h>

typedef __attribute__((ext_vector_type(16))) _Float16 v16h;
typedef __attribute__((ext_vector_type(8)))  _Float16 v8h;
typedef __attribute__((ext_vector_type(4)))  _Float16 v4h;
typedef __attribute__((ext_vector_type(8)))  float    v8f;
typedef __attribute__((ext_vector_type(4)))  unsigned v4u;
typedef __attribute__((ext_vector_type(8)))  unsigned v8u;

#define B_SZ   16
#define N_PTS  1024
#define KNN    20
#define NEG_BIG (-3.0e38f)

// ---------- WMMA helpers (gfx1250 wave32 layouts, CDNA5 ISA) ----------
// 16-bit A/B fragment element e (0..15), lane-half h: K = (e&7) + 16*(e>>3) + 8*h.
// => elements 0..7 are contiguous at K = kbase+8*h, elements 8..15 at +16.
__device__ __forceinline__ int kmap(int e, int half) {
  return (e & 7) + ((e >> 3) << 4) + (half << 3);
}

union frag16 { v16h f; v8h h[2]; };

// lane holds row `row` of a row-major [rows][stride] f16 matrix: two b128 loads
__device__ __forceinline__ v16h frag_row_vec(const _Float16* base, int stride, int row,
                                             int kbase, int half) {
  const _Float16* p = base + (size_t)row * stride + kbase + half * 8;
  frag16 u;
  u.h[0] = *(const v8h*)p;
  u.h[1] = *(const v8h*)(p + 16);
  return u.f;
}

__device__ __forceinline__ v8f wmma_f16(v16h a, v16h b, v8f c) {
  return __builtin_amdgcn_wmma_f32_16x16x32_f16(false, a, false, b, (short)0, c,
                                                false, false);
}

__device__ __forceinline__ void topk_insert(float* bd, int* bi, float d, int j) {
  if (d < bd[KNN - 1]) {
    int p = KNN - 1;
    while (p > 0 && bd[p - 1] > d) { bd[p] = bd[p - 1]; bi[p] = bi[p - 1]; --p; }
    bd[p] = d; bi[p] = j;
  }
}

// ---------- 0. weight prep: fold concat-split weights, f16 + TRANSPOSE ----------
// Wc1  f32 [3][128] : cols 0..63 = W1a_top - W1a_bot, cols 64..127 = W1a_bot
// W1bT/W1cT f16 [64(N)][64(K)]   (transposed so B-frags are contiguous loads)
// Wc2T f16 [256(N)][64(K)] : rows 0..127 = (W2_top-W2_bot)^T, 128..255 = W2_bot^T
// WlT  f16 [1024(N)][192(K)]
__global__ void prep_kernel(const float* W1a, const float* W1b, const float* W1c,
                            const float* W2, const float* Wl,
                            float* Wc1, _Float16* W1bT, _Float16* W1cT,
                            _Float16* Wc2T, _Float16* WlT) {
  int t = blockIdx.x * 256 + threadIdx.x;
  if (t < 3 * 128) {
    int d = t >> 7, c = t & 127;
    Wc1[t] = (c < 64) ? (W1a[d * 64 + c] - W1a[(d + 3) * 64 + c])
                      : W1a[(d + 3) * 64 + (c - 64)];
  }
  if (t < 64 * 64) {
    int n = t >> 6, k = t & 63;
    W1bT[t] = (_Float16)W1b[k * 64 + n];
    W1cT[t] = (_Float16)W1c[k * 64 + n];
  }
  if (t < 256 * 64) {
    int n = t >> 6, k = t & 63;
    float v = (n < 128) ? (W2[k * 128 + n] - W2[(k + 64) * 128 + n])
                        : W2[(k + 64) * 128 + (n - 128)];
    Wc2T[t] = (_Float16)v;
  }
  if (t < 1024 * 192) {
    int n = t / 192, k = t - n * 192;
    WlT[t] = (_Float16)Wl[k * 1024 + n];
  }
}

// ---------- 1. kNN on pos (D=3); batch staged into LDS by the Tensor Data Mover ---
__global__ __launch_bounds__(256) void knn1_kernel(const float* pos, int* idx1) {
  __shared__ float P[N_PTS * 3];          // only LDS object => offset 0
  int b = blockIdx.x >> 2;
  int qbase = (blockIdx.x & 3) * 256;
  if (threadIdx.x < 32) {                 // wave 0 issues one TDM 1-D tile load
    unsigned long long ga = (unsigned long long)(pos + (size_t)b * N_PTS * 3);
    // D# group0: count=1 | lds_addr=0 | global_addr | type=2
    v4u g0 = {1u, 0u, (unsigned)ga, (unsigned)(ga >> 32) | (2u << 30)};
    // D# group1: data_size=4B; tensor_dim0=tile_dim0=3072; stride0=3072
    v8u g1 = {(2u << 16),        // [17:16] data_size = 2 (4 bytes)
              (3072u << 16),     // [63:48] tensor_dim0 lo16
              0u,                // tensor_dim0 hi / tensor_dim1
              (3072u << 16),     // [127:112] tile_dim0
              0u,                // tile_dim1/2 unused
              3072u,             // [207:160] tensor_dim0_stride lo32
              0u, 0u};
    asm volatile("tensor_load_to_lds %0, %1" :: "s"(g0), "s"(g1) : "memory");
    __builtin_amdgcn_s_wait_tensorcnt(0);
  }
  __syncthreads();

  int q = qbase + threadIdx.x;
  float qx = P[q * 3 + 0], qy = P[q * 3 + 1], qz = P[q * 3 + 2];
  float bd[KNN]; int bi[KNN];
#pragma unroll
  for (int t = 0; t < KNN; ++t) { bd[t] = 1e30f; bi[t] = 0; }
  for (int j = 0; j < N_PTS; ++j) {
    if (j == q) continue;
    float dx = qx - P[j * 3 + 0], dy = qy - P[j * 3 + 1], dz = qz - P[j * 3 + 2];
    topk_insert(bd, bi, dx * dx + dy * dy + dz * dz, j);
  }
  for (int t = 0; t < KNN; ++t) idx1[(b * N_PTS + q) * KNN + t] = bi[t];
}

// ---------- 2. EdgeConv1 fully fused: h1 on-the-fly, 2 WMMA layers, max over k ----
__global__ __launch_bounds__(32) void edgeconv1_kernel(
    const float* pos, const int* idx1, const float* Wc1, const float* b1a_,
    const _Float16* W1bT, const _Float16* W1cT, const float* b1b_,
    const float* b1c_, _Float16* feat) {
  int b = blockIdx.x >> 6, pt = blockIdx.x & 63;
  int rowbase = b * N_PTS + pt * 16;
  int lane = threadIdx.x, r = lane & 15, half = lane >> 4;

  v16h B2[4][2], B3[4][2];
#pragma unroll
  for (int nt = 0; nt < 4; ++nt)
#pragma unroll
    for (int kc = 0; kc < 2; ++kc) {
      B2[nt][kc] = frag_row_vec(W1bT, 64, nt * 16 + r, kc * 32, half);
      B3[nt][kc] = frag_row_vec(W1cT, 64, nt * 16 + r, kc * 32, half);
    }
  float bb2[4], bb3[4];
#pragma unroll
  for (int nt = 0; nt < 4; ++nt) { bb2[nt] = b1b_[nt * 16 + r]; bb3[nt] = b1c_[nt * 16 + r]; }

  int qi = rowbase + r;
  float qx = pos[qi * 3 + 0], qy = pos[qi * 3 + 1], qz = pos[qi * 3 + 2];

  v8f vmax[4];
#pragma unroll
  for (int nt = 0; nt < 4; ++nt)
#pragma unroll
    for (int v = 0; v < 8; ++v) vmax[nt][v] = NEG_BIG;

  __shared__ _Float16 h2s[16][64];

  for (int t = 0; t < KNN; ++t) {
    int j = idx1[qi * KNN + t];
    int jg = b * N_PTS + j;
    float nx = pos[jg * 3 + 0], ny = pos[jg * 3 + 1], nz = pos[jg * 3 + 2];

    // layer 1: h1 = relu(xi@(Wtop-Wbot) + xj@Wbot + b1a) straight into A frags
    v16h A[2];
#pragma unroll
    for (int kc = 0; kc < 2; ++kc)
#pragma unroll
      for (int e = 0; e < 16; ++e) {
        int c = kc * 32 + kmap(e, half);
        float v = b1a_[c];
        v = fmaf(qx, Wc1[0 * 128 + c], v);
        v = fmaf(qy, Wc1[1 * 128 + c], v);
        v = fmaf(qz, Wc1[2 * 128 + c], v);
        v = fmaf(nx, Wc1[0 * 128 + 64 + c], v);
        v = fmaf(ny, Wc1[1 * 128 + 64 + c], v);
        v = fmaf(nz, Wc1[2 * 128 + 64 + c], v);
        A[kc][e] = (_Float16)fmaxf(v, 0.0f);
      }

    // layer 2: 64->64 WMMA, bias folded into C init
    v8f c2[4];
#pragma unroll
    for (int nt = 0; nt < 4; ++nt) {
#pragma unroll
      for (int v = 0; v < 8; ++v) c2[nt][v] = bb2[nt];
#pragma unroll
      for (int kc = 0; kc < 2; ++kc) c2[nt] = wmma_f16(A[kc], B2[nt][kc], c2[nt]);
    }
    __syncthreads();
#pragma unroll
    for (int nt = 0; nt < 4; ++nt)
#pragma unroll
      for (int v = 0; v < 8; ++v)
        h2s[v + 8 * half][nt * 16 + r] = (_Float16)fmaxf(c2[nt][v], 0.0f);
    __syncthreads();

    // layer 3: 64->64 WMMA (no relu), then running max over neighbors
    v16h A3[2];
#pragma unroll
    for (int kc = 0; kc < 2; ++kc) {
      const _Float16* hp = &h2s[r][kc * 32 + half * 8];
      frag16 u;
      u.h[0] = *(const v8h*)hp;
      u.h[1] = *(const v8h*)(hp + 16);
      A3[kc] = u.f;
    }
    v8f c3[4];
#pragma unroll
    for (int nt = 0; nt < 4; ++nt) {
#pragma unroll
      for (int v = 0; v < 8; ++v) c3[nt][v] = bb3[nt];
#pragma unroll
      for (int kc = 0; kc < 2; ++kc) c3[nt] = wmma_f16(A3[kc], B3[nt][kc], c3[nt]);
#pragma unroll
      for (int v = 0; v < 8; ++v) vmax[nt][v] = fmaxf(vmax[nt][v], c3[nt][v]);
    }
  }
#pragma unroll
  for (int nt = 0; nt < 4; ++nt)
#pragma unroll
    for (int v = 0; v < 8; ++v)
      feat[(size_t)(rowbase + v + 8 * half) * 192 + nt * 16 + r] = (_Float16)vmax[nt][v];
}

// ---------- 3. squared norms of x1 (vectorized b128 reads) ----------
__global__ void sqnorm_kernel(const _Float16* feat, float* sq) {
  int i = blockIdx.x * 256 + threadIdx.x;  // < B*N
  const v8h* row = (const v8h*)(feat + (size_t)i * 192);
  float s = 0.0f;
#pragma unroll
  for (int cv = 0; cv < 8; ++cv) {
    v8h v = row[cv];
#pragma unroll
    for (int e = 0; e < 8; ++e) { float x = (float)v[e]; s += x * x; }
  }
  sq[i] = s;
}

// ---------- 4. kNN on x1: WMMA Gram stripes + fused top-20 ----------
__global__ __launch_bounds__(32) void knn2_kernel(const _Float16* feat, const float* sq,
                                                  int* idx2) {
  __shared__ float dist[16][512];
  int b = blockIdx.x >> 6, pt = blockIdx.x & 63;
  int rowbase = b * N_PTS + pt * 16;
  int lane = threadIdx.x, r = lane & 15, half = lane >> 4;

  v16h A[2];
#pragma unroll
  for (int kc = 0; kc < 2; ++kc) A[kc] = frag_row_vec(feat, 192, rowbase + r, kc * 32, half);

  float bd[KNN]; int bi[KNN];
#pragma unroll
  for (int t = 0; t < KNN; ++t) { bd[t] = 1e30f; bi[t] = 0; }

  for (int pass = 0; pass < 2; ++pass) {
    for (int jt = 0; jt < 32; ++jt) {
      int jb = (pass * 32 + jt) * 16;                  // within-batch col base
      v16h Bf[2];
#pragma unroll
      for (int kc = 0; kc < 2; ++kc)
        Bf[kc] = frag_row_vec(feat, 192, b * N_PTS + jb + r, kc * 32, half);
      v8f c;
#pragma unroll
      for (int v = 0; v < 8; ++v) c[v] = 0.0f;
      c = wmma_f16(A[0], Bf[0], c);
      c = wmma_f16(A[1], Bf[1], c);
      int colg = jb + r;
      float sqj = sq[b * N_PTS + colg];
#pragma unroll
      for (int v = 0; v < 8; ++v) {
        int row = v + 8 * half;
        float d = sq[rowbase + row] + sqj - 2.0f * c[v];
        if (pt * 16 + row == colg) d = 1e30f;          // exclude self
        dist[row][jt * 16 + r] = d;
      }
    }
    __syncthreads();
    if (lane < 16) {
      for (int cc = 0; cc < 512; ++cc)
        topk_insert(bd, bi, dist[lane][cc], pass * 512 + cc);
    }
    __syncthreads();
  }
  if (lane < 16)
    for (int t = 0; t < KNN; ++t) idx2[(rowbase + lane) * KNN + t] = bi[t];
}

// ---------- 5. CY2 = x1 @ [W2top-W2bot | W2bot]  (B*N x 256, f32) ----------
__global__ __launch_bounds__(32) void cy2_kernel(const _Float16* feat,
                                                 const _Float16* Wc2T, float* CY2) {
  int rowbase = blockIdx.x * 16;
  int lane = threadIdx.x, r = lane & 15, half = lane >> 4;
  v16h A[2];
#pragma unroll
  for (int kc = 0; kc < 2; ++kc) A[kc] = frag_row_vec(feat, 192, rowbase + r, kc * 32, half);
  for (int nt = 0; nt < 16; ++nt) {
    v16h Bf[2];
#pragma unroll
    for (int kc = 0; kc < 2; ++kc) Bf[kc] = frag_row_vec(Wc2T, 64, nt * 16 + r, kc * 32, half);
    v8f c;
#pragma unroll
    for (int v = 0; v < 8; ++v) c[v] = 0.0f;
    c = wmma_f16(A[0], Bf[0], c);
    c = wmma_f16(A[1], Bf[1], c);
#pragma unroll
    for (int v = 0; v < 8; ++v)
      CY2[(size_t)(rowbase + v + 8 * half) * 256 + nt * 16 + r] = c[v];
  }
}

// ---------- 6. x2[i] = C2[i] + b2 + max_j Y2[j]; write feat[:,64:192] ----------
__global__ void x2_kernel(const float* CY2, const int* idx2, const float* b2_,
                          _Float16* feat) {
  int t = blockIdx.x * 256 + threadIdx.x;  // < B*N*32
  int fq = t & 31, i = t >> 5, b = i >> 10;
  int f = fq * 4;
  float4 m = make_float4(NEG_BIG, NEG_BIG, NEG_BIG, NEG_BIG);
  for (int n = 0; n < KNN; ++n) {
    int j = idx2[i * KNN + n];
    float4 y = *(const float4*)(CY2 + (size_t)(b * N_PTS + j) * 256 + 128 + f);
    m.x = fmaxf(m.x, y.x); m.y = fmaxf(m.y, y.y);
    m.z = fmaxf(m.z, y.z); m.w = fmaxf(m.w, y.w);
  }
  float4 c  = *(const float4*)(CY2 + (size_t)i * 256 + f);
  float4 bb = *(const float4*)(b2_ + f);
  v4h o;
  o[0] = (_Float16)(c.x + bb.x + m.x);
  o[1] = (_Float16)(c.y + bb.y + m.y);
  o[2] = (_Float16)(c.z + bb.z + m.z);
  o[3] = (_Float16)(c.w + bb.w + m.w);
  *(v4h*)(feat + (size_t)i * 192 + 64 + f) = o;
}

// ---------- 7. lin1 + global max pool fused ----------
__global__ __launch_bounds__(32) void lin1_kernel(const _Float16* feat,
                                                  const _Float16* WlT,
                                                  const float* bl_, float* g) {
  int b = blockIdx.x >> 6, ct = blockIdx.x & 63;
  int lane = threadIdx.x, r = lane & 15, half = lane >> 4;
  v16h Bf[6];
#pragma unroll
  for (int kc = 0; kc < 6; ++kc) Bf[kc] = frag_row_vec(WlT, 192, ct * 16 + r, kc * 32, half);
  float colmax = NEG_BIG;
  for (int rt = 0; rt < 64; ++rt) {
    int rowbase = b * N_PTS + rt * 16;
    if (rt + 1 < 64) __builtin_prefetch(feat + (size_t)(rowbase + 16) * 192, 0, 0);
    v8f c;
#pragma unroll
    for (int v = 0; v < 8; ++v) c[v] = 0.0f;
#pragma unroll
    for (int kc = 0; kc < 6; ++kc) {
      v16h A = frag_row_vec(feat, 192, rowbase + r, kc * 32, half);
      c = wmma_f16(A, Bf[kc], c);
    }
    float m = c[0];
#pragma unroll
    for (int v = 1; v < 8; ++v) m = fmaxf(m, c[v]);
    colmax = fmaxf(colmax, m);
  }
  colmax = fmaxf(colmax, __shfl_xor(colmax, 16, 32));   // merge C halves (rows 0-7/8-15)
  if (lane < 16) g[b * 1024 + ct * 16 + lane] = colmax + bl_[ct * 16 + lane];
}

// ---------- 8. head (tiny, plain VALU) ----------
__global__ void head1_kernel(const float* g, const float* Wm1, const float* bm1,
                             float* h1) {
  int t = blockIdx.x * 256 + threadIdx.x;  // < 16*512
  int rt = t >> 9, c = t & 511;
  float acc = bm1[c];
  for (int k = 0; k < 1024; ++k) acc = fmaf(g[rt * 1024 + k], Wm1[k * 512 + c], acc);
  h1[t] = fmaxf(acc, 0.0f);
}
__global__ void head2_kernel(const float* h1, const float* Wm2, const float* bm2,
                             float* h2) {
  int t = blockIdx.x * 256 + threadIdx.x;  // < 16*256
  int rt = t >> 8, c = t & 255;
  float acc = bm2[c];
  for (int k = 0; k < 512; ++k) acc = fmaf(h1[rt * 512 + k], Wm2[k * 256 + c], acc);
  h2[t] = fmaxf(acc, 0.0f);
}
__global__ __launch_bounds__(640) void head3_kernel(const float* h2, const float* Wm3,
                                                    const float* bm3, float* out) {
  __shared__ float lg[16][40];
  __shared__ float mrow[16], srow[16];
  int t = threadIdx.x;            // 0..639
  int rt = t / 40, c = t % 40;
  float acc = bm3[c];
  for (int k = 0; k < 256; ++k) acc = fmaf(h2[rt * 256 + k], Wm3[k * 40 + c], acc);
  lg[rt][c] = acc;
  __syncthreads();
  if (t < 16) {
    float m = NEG_BIG;
    for (int cc = 0; cc < 40; ++cc) m = fmaxf(m, lg[t][cc]);
    float s = 0.0f;
    for (int cc = 0; cc < 40; ++cc) s += expf(lg[t][cc] - m);
    mrow[t] = m; srow[t] = logf(s);
  }
  __syncthreads();
  out[rt * 40 + c] = lg[rt][c] - mrow[rt] - srow[rt];
}

// ---------- launcher ----------
extern "C" void kernel_launch(void* const* d_in, const int* in_sizes, int n_in,
                              void* d_out, int out_size, void* d_ws, size_t ws_size,
                              hipStream_t stream) {
  (void)in_sizes; (void)n_in; (void)out_size; (void)ws_size;
  const float* pos  = (const float*)d_in[0];
  const float* W1a  = (const float*)d_in[1];
  const float* b1a  = (const float*)d_in[2];
  const float* W1b  = (const float*)d_in[3];
  const float* b1b  = (const float*)d_in[4];
  const float* W1c  = (const float*)d_in[5];
  const float* b1c  = (const float*)d_in[6];
  const float* W2   = (const float*)d_in[7];
  const float* b2   = (const float*)d_in[8];
  const float* Wl   = (const float*)d_in[9];
  const float* bl   = (const float*)d_in[10];
  const float* Wm1  = (const float*)d_in[11];
  const float* bm1  = (const float*)d_in[12];
  const float* Wm2  = (const float*)d_in[13];
  const float* bm2  = (const float*)d_in[14];
  const float* Wm3  = (const float*)d_in[15];
  const float* bm3  = (const float*)d_in[16];
  float* out = (float*)d_out;

  char* w = (char*)d_ws;
  size_t off = 0;
  auto alloc = [&](size_t bytes) -> void* {
    off = (off + 255) & ~(size_t)255;
    void* p = w + off;
    off += bytes;
    return p;
  };
  int*       idx1  = (int*)alloc((size_t)B_SZ * N_PTS * KNN * 4);
  int*       idx2  = (int*)alloc((size_t)B_SZ * N_PTS * KNN * 4);
  float*     Wc1   = (float*)alloc(3 * 128 * 4);
  _Float16*  W1bT  = (_Float16*)alloc(64 * 64 * 2);
  _Float16*  W1cT  = (_Float16*)alloc(64 * 64 * 2);
  _Float16*  Wc2T  = (_Float16*)alloc(256 * 64 * 2);
  _Float16*  WlT   = (_Float16*)alloc(1024 * 192 * 2);
  _Float16*  feat  = (_Float16*)alloc((size_t)B_SZ * N_PTS * 192 * 2);
  float*     sq    = (float*)alloc((size_t)B_SZ * N_PTS * 4);
  float*     CY2   = (float*)alloc((size_t)B_SZ * N_PTS * 256 * 4);
  float*     g     = (float*)alloc((size_t)B_SZ * 1024 * 4);
  float*     h1    = (float*)alloc((size_t)B_SZ * 512 * 4);
  float*     h2    = (float*)alloc((size_t)B_SZ * 256 * 4);

  prep_kernel<<<(192 * 1024 + 255) / 256, 256, 0, stream>>>(
      W1a, W1b, W1c, W2, Wl, Wc1, W1bT, W1cT, Wc2T, WlT);
  knn1_kernel<<<B_SZ * 4, 256, 0, stream>>>(pos, idx1);
  edgeconv1_kernel<<<B_SZ * 64, 32, 0, stream>>>(
      pos, idx1, Wc1, b1a, W1bT, W1cT, b1b, b1c, feat);
  sqnorm_kernel<<<(B_SZ * N_PTS) / 256, 256, 0, stream>>>(feat, sq);
  knn2_kernel<<<B_SZ * 64, 32, 0, stream>>>(feat, sq, idx2);
  cy2_kernel<<<(B_SZ * N_PTS) / 16, 32, 0, stream>>>(feat, Wc2T, CY2);
  x2_kernel<<<(B_SZ * N_PTS * 32) / 256, 256, 0, stream>>>(CY2, idx2, b2, feat);
  lin1_kernel<<<B_SZ * 64, 32, 0, stream>>>(feat, WlT, bl, g);
  head1_kernel<<<(B_SZ * 512) / 256, 256, 0, stream>>>(g, Wm1, bm1, h1);
  head2_kernel<<<(B_SZ * 256) / 256, 256, 0, stream>>>(h1, Wm2, bm2, h2);
  head3_kernel<<<1, 640, 0, stream>>>(h2, Wm3, bm3, out);
}
